// MambaBlock_10892037062805
// MI455X (gfx1250) — matrix operand order repaired and checked by reading, not compiled, not run on hardware
//
#include <hip/hip_runtime.h>
#include <hip/hip_bf16.h>
#include <cstddef>

// ---------------- problem constants ----------------
#define BSZ   16
#define LSEQ  2048
#define DMODEL 256
#define NSTATE 16
#define RRANK  16
#define NCHUNK 32
#define CLEN   64          // LSEQ / NCHUNK
#define NTOK   (BSZ*LSEQ)  // 32768
#define MT     4           // M-tiles per wave in WMMA GEMMs (B-fragment reuse)

typedef __attribute__((ext_vector_type(16))) __bf16 v16bf;
typedef __attribute__((ext_vector_type(8)))  float  v8f;

// ---------------- WMMA fragment builders (bf16 16x16x32) ----------------
// A fragment: 16x32 (MxK), row-major source with leading dim lda (floats).
// ISA 7.12.2: lane m = lane&15 ; koff = 8*(lane>=16)
//   elements 0..7  -> K = kb + koff + e
//   elements 8..15 -> K = kb + koff + 16 + (e-8)
__device__ __forceinline__ v16bf frag_a_f32(const float* __restrict__ A, int lda,
                                            int kb, int lane) {
  v16bf a;
  int m    = lane & 15;
  int koff = (lane >> 4) << 3;
  const float* r = A + (size_t)m * lda + kb + koff;
#pragma unroll
  for (int e = 0; e < 8; ++e) a[e]     = (__bf16)r[e];
#pragma unroll
  for (int e = 0; e < 8; ++e) a[e + 8] = (__bf16)r[16 + e];
  return a;
}

// B fragment: 32x16 (KxN), row-major source with leading dim ldb (floats).
// lane n = lane&15 ; K = kb + 8*(lane>=16) + e (e<8), kb + 8*(lane>=16) + 16 + (e-8) (e>=8)
__device__ __forceinline__ v16bf frag_b_f32(const float* __restrict__ B, int ldb,
                                            int kb, int lane) {
  v16bf b;
  int n  = lane & 15;
  int ks = (lane >> 4) << 3;
  const float* p = B + (size_t)(kb + ks) * ldb + n;
#pragma unroll
  for (int e = 0; e < 16; ++e) b[e] = (__bf16)p[(size_t)(e < 8 ? e : e + 8) * ldb];
  return b;
}

// ---------------- kernel 1: LayerNorm (one wave per token) ----------------
__global__ void k_layernorm(const float* __restrict__ x, const float* __restrict__ w,
                            const float* __restrict__ bb, float* __restrict__ out) {
  int lane  = threadIdx.x & 31;
  int token = blockIdx.x * 8 + (threadIdx.x >> 5);
  const float* xp = x + (size_t)token * DMODEL;
  float v[8];
  float s = 0.f;
#pragma unroll
  for (int i = 0; i < 8; ++i) { v[i] = xp[i * 32 + lane]; s += v[i]; }
#pragma unroll
  for (int m = 16; m >= 1; m >>= 1) s += __shfl_xor(s, m, 32);
  float mu = s * (1.0f / DMODEL);
  float q = 0.f;
#pragma unroll
  for (int i = 0; i < 8; ++i) { float dv = v[i] - mu; q += dv * dv; }
#pragma unroll
  for (int m = 16; m >= 1; m >>= 1) q += __shfl_xor(q, m, 32);
  float rstd = rsqrtf(q * (1.0f / DMODEL) + 1e-5f);
  float* op = out + (size_t)token * DMODEL;
#pragma unroll
  for (int i = 0; i < 8; ++i) {
    int ch = i * 32 + lane;
    op[ch] = (v[i] - mu) * rstd * w[ch] + bb[ch];
  }
}

// -------- kernel 2: x_proj GEMM (32768 x 48 x 256), WMMA, 4 M-tiles/wave --------
__global__ void k_xproj(const float* __restrict__ lnx, const float* __restrict__ W,
                        float* __restrict__ dbl) {
  int lane  = threadIdx.x & 31;
  int tile  = blockIdx.x * 8 + (threadIdx.x >> 5);
  int mg    = tile / 3;        // group of MT token tiles
  int ctile = tile % 3;        // 16-column tile of 48
  v8f acc[MT];
#pragma unroll
  for (int t = 0; t < MT; ++t) acc[t] = (v8f){};
  for (int kb = 0; kb < DMODEL; kb += 32) {
    v16bf b = frag_b_f32(W + ctile * 16, 48, kb, lane);
#pragma unroll
    for (int t = 0; t < MT; ++t) {
      const float* A = lnx + (size_t)(mg * MT + t) * 16 * DMODEL;
      v16bf a = frag_a_f32(A, DMODEL, kb, lane);
      acc[t] = __builtin_amdgcn_wmma_f32_16x16x32_bf16(false, a, false, b, (short)0,
                                                       acc[t], false, false);
    }
  }
  int n = lane & 15;
#pragma unroll
  for (int t = 0; t < MT; ++t) {
    int ttile = mg * MT + t;
#pragma unroll
    for (int g = 0; g < 8; ++g) {
      int m = g + ((lane >> 4) << 3);
      dbl[(size_t)(ttile * 16 + m) * 48 + ctile * 16 + n] = acc[t][g];
    }
  }
}

// -- kernel 3: dt_proj GEMM (32768 x 256 x 16, K padded) + softplus, 4 M-tiles --
__global__ void k_dtproj(const float* __restrict__ dbl, const float* __restrict__ W,
                         const float* __restrict__ bias, float* __restrict__ delta) {
  int lane  = threadIdx.x & 31;
  int tile  = blockIdx.x * 8 + (threadIdx.x >> 5);
  int mg    = tile >> 4;       // group of MT token tiles
  int ctile = tile & 15;       // 16-column tile of 256
  v16bf b;
  {
    int n  = lane & 15;
    int ks = (lane >> 4) << 3;                         // 0 or 8
#pragma unroll
    for (int e = 0; e < 16; ++e) {
      int k = ks + (e < 8 ? e : e + 8);
      b[e] = (k < RRANK) ? (__bf16)W[(size_t)k * DMODEL + ctile * 16 + n]
                         : (__bf16)0.0f;
    }
  }
  int n = lane & 15;
#pragma unroll
  for (int t = 0; t < MT; ++t) {
    int ttile = mg * MT + t;
    v16bf a;
    {
      int m    = lane & 15;
      int koff = (lane >> 4) << 3;                     // 0 or 8 (< 16 always)
      const float* r = dbl + (size_t)(ttile * 16 + m) * 48 + koff;
#pragma unroll
      for (int e = 0; e < 8; ++e) a[e]     = (__bf16)r[e];   // K = koff+e < 16
#pragma unroll
      for (int e = 0; e < 8; ++e) a[e + 8] = (__bf16)0.0f;   // K >= 16 padded
    }
    v8f acc = {};
    acc = __builtin_amdgcn_wmma_f32_16x16x32_bf16(false, a, false, b, (short)0, acc,
                                                  false, false);
#pragma unroll
    for (int g = 0; g < 8; ++g) {
      int m   = g + ((lane >> 4) << 3);
      int col = ctile * 16 + n;
      float v = acc[g] + bias[col];
      float sp = (v > 20.0f) ? v : log1pf(expf(v));    // softplus
      delta[(size_t)(ttile * 16 + m) * DMODEL + col] = sp;
    }
  }
}

// ---- kernel 4: scan pass A — per-chunk local scan (h0=0) + decay product ----
__global__ void k_scanA(const float* __restrict__ delta, const float* __restrict__ lnx,
                        const float* __restrict__ dbl,   const float* __restrict__ A_log,
                        float* __restrict__ cP, float* __restrict__ cH) {
  int g   = blockIdx.x * blockDim.x + threadIdx.x;     // 131072 = b*8192 + c*256 + d
  int dch = g & 255;
  int c   = (g >> 8) & (NCHUNK - 1);
  int b   = g >> 13;
  float Ar[NSTATE], h[NSTATE], P[NSTATE];
#pragma unroll
  for (int n = 0; n < NSTATE; ++n) {
    Ar[n] = -__expf(A_log[dch * NSTATE + n]);
    h[n] = 0.f; P[n] = 1.f;
  }
  int t0 = c * CLEN;
  for (int s = 0; s < CLEN; ++s) {
    size_t tok = (size_t)b * LSEQ + t0 + s;
    float del = delta[tok * DMODEL + dch];
    float xv  = lnx[tok * DMODEL + dch];
    const float* bv = dbl + tok * 48 + RRANK;
    float dx = del * xv;
#pragma unroll
    for (int n = 0; n < NSTATE; ++n) {
      float dA = __expf(del * Ar[n]);
      h[n] = dA * h[n] + dx * bv[n];
      P[n] *= dA;
    }
  }
  size_t base = (size_t)g * NSTATE;                    // [b][c][d][n]
#pragma unroll
  for (int n = 0; n < NSTATE; ++n) { cP[base + n] = P[n]; cH[base + n] = h[n]; }
}

// ---- kernel 5: scan pass B — carry prefix across the 32 chunks ----
__global__ void k_scanB(const float* __restrict__ cP, const float* __restrict__ cH,
                        float* __restrict__ cH0) {
  int g   = blockIdx.x * blockDim.x + threadIdx.x;     // 65536 = b*4096 + d*16 + n
  int n   = g & 15;
  int dch = (g >> 4) & 255;
  int b   = g >> 12;
  float h0 = 0.f;
  for (int c = 0; c < NCHUNK; ++c) {
    size_t idx = ((((size_t)b * NCHUNK + c) * DMODEL + dch) * NSTATE) + n;
    cH0[idx] = h0;                                     // state entering chunk c
    h0 = cP[idx] * h0 + cH[idx];
  }
}

// ---- kernel 6: scan pass C — replay with correct h0, emit gelu(y + x*D) ----
__global__ void k_scanC(const float* __restrict__ delta, const float* __restrict__ lnx,
                        const float* __restrict__ dbl,   const float* __restrict__ A_log,
                        const float* __restrict__ Dsk,   const float* __restrict__ cH0,
                        float* __restrict__ yg) {
  int g   = blockIdx.x * blockDim.x + threadIdx.x;
  int dch = g & 255;
  int c   = (g >> 8) & (NCHUNK - 1);
  int b   = g >> 13;
  float Ar[NSTATE], h[NSTATE];
  size_t cbase = (size_t)g * NSTATE;
#pragma unroll
  for (int n = 0; n < NSTATE; ++n) {
    Ar[n] = -__expf(A_log[dch * NSTATE + n]);
    h[n] = cH0[cbase + n];
  }
  float Dv = Dsk[dch];
  int t0 = c * CLEN;
  for (int s = 0; s < CLEN; ++s) {
    size_t tok = (size_t)b * LSEQ + t0 + s;
    float del = delta[tok * DMODEL + dch];
    float xv  = lnx[tok * DMODEL + dch];
    const float* bv = dbl + tok * 48 + RRANK;
    const float* cv = dbl + tok * 48 + RRANK + NSTATE;
    float dx = del * xv;
    float y = 0.f;
#pragma unroll
    for (int n = 0; n < NSTATE; ++n) {
      float dA = __expf(del * Ar[n]);
      h[n] = dA * h[n] + dx * bv[n];
      y += h[n] * cv[n];
    }
    float o = y + xv * Dv;
    float ge = 0.5f * o * (1.0f + erff(o * 0.70710678118f));  // exact gelu
    yg[tok * DMODEL + dch] = ge;
  }
}

// ---- kernel 7: GLU GEMM (both halves, 4 M-tiles/wave) + gate + residual ----
__global__ void k_glu(const float* __restrict__ yg, const float* __restrict__ W,
                      const float* __restrict__ bias, const float* __restrict__ xin,
                      float* __restrict__ out) {
  int lane  = threadIdx.x & 31;
  int tile  = blockIdx.x * 8 + (threadIdx.x >> 5);
  int mg    = tile >> 4;                               // group of MT token tiles
  int ctile = tile & 15;                               // 16 cols of first half
  v8f acc1[MT], acc2[MT];
#pragma unroll
  for (int t = 0; t < MT; ++t) { acc1[t] = (v8f){}; acc2[t] = (v8f){}; }
  for (int kb = 0; kb < DMODEL; kb += 32) {
    // prefetch next K-block of weights into cache (global_prefetch_b8)
    if (kb + 32 < DMODEL) {
      __builtin_prefetch(W + (size_t)(kb + 32) * 512 + ctile * 16 + (lane & 15), 0, 1);
      __builtin_prefetch(W + (size_t)(kb + 32) * 512 + DMODEL + ctile * 16 + (lane & 15), 0, 1);
    }
    v16bf b1 = frag_b_f32(W + ctile * 16,          512, kb, lane);
    v16bf b2 = frag_b_f32(W + DMODEL + ctile * 16, 512, kb, lane);
#pragma unroll
    for (int t = 0; t < MT; ++t) {
      const float* A = yg + (size_t)(mg * MT + t) * 16 * DMODEL;
      v16bf a = frag_a_f32(A, DMODEL, kb, lane);
      acc1[t] = __builtin_amdgcn_wmma_f32_16x16x32_bf16(false, a, false, b1, (short)0,
                                                        acc1[t], false, false);
      acc2[t] = __builtin_amdgcn_wmma_f32_16x16x32_bf16(false, a, false, b2, (short)0,
                                                        acc2[t], false, false);
    }
  }
  int n = lane & 15;
#pragma unroll
  for (int t = 0; t < MT; ++t) {
    int ttile = mg * MT + t;
#pragma unroll
    for (int g = 0; g < 8; ++g) {
      int m   = g + ((lane >> 4) << 3);
      int col = ctile * 16 + n;
      float a1 = acc1[t][g] + bias[col];
      float a2 = acc2[t][g] + bias[col + DMODEL];
      float sg = 1.0f / (1.0f + __expf(-a2));
      size_t tok = (size_t)ttile * 16 + m;
      out[tok * DMODEL + col] = a1 * sg + xin[tok * DMODEL + col];
    }
  }
}

// ---------------- launcher ----------------
extern "C" void kernel_launch(void* const* d_in, const int* in_sizes, int n_in,
                              void* d_out, int out_size, void* d_ws, size_t ws_size,
                              hipStream_t stream) {
  (void)in_sizes; (void)n_in; (void)out_size; (void)ws_size;
  const float* x        = (const float*)d_in[0];
  const float* ln_w     = (const float*)d_in[1];
  const float* ln_b     = (const float*)d_in[2];
  const float* x_proj_w = (const float*)d_in[3];
  const float* dt_proj_w= (const float*)d_in[4];
  const float* dt_proj_b= (const float*)d_in[5];
  const float* A_log    = (const float*)d_in[6];
  const float* D_skip   = (const float*)d_in[7];
  const float* glu_w    = (const float*)d_in[8];
  const float* glu_b    = (const float*)d_in[9];
  float* out = (float*)d_out;

  // workspace partition (floats)
  float* ws    = (float*)d_ws;
  float* lnx   = ws;                              // NTOK*DMODEL   =  8,388,608
  float* dbl   = lnx   + (size_t)NTOK * DMODEL;   // NTOK*48       =  1,572,864
  float* delta = dbl   + (size_t)NTOK * 48;       // NTOK*DMODEL   =  8,388,608
  float* yg    = delta + (size_t)NTOK * DMODEL;   // NTOK*DMODEL   =  8,388,608
  float* cP    = yg    + (size_t)NTOK * DMODEL;   // B*C*D*N       =  2,097,152
  float* cH    = cP    + (size_t)BSZ * NCHUNK * DMODEL * NSTATE;
  float* cH0   = cH    + (size_t)BSZ * NCHUNK * DMODEL * NSTATE;

  const int MTILES = NTOK / 16;                   // 2048 token tiles
  const int MGRP   = MTILES / MT;                 // 512 tile groups

  // 1) LayerNorm: 32768 tokens, 8 waves (tokens) per 256-thread block
  k_layernorm<<<NTOK / 8, 256, 0, stream>>>(x, ln_w, ln_b, lnx);
  // 2) x_proj: 512 groups * 3 col tiles = 1536 waves, 8 per block
  k_xproj<<<MGRP * 3 / 8, 256, 0, stream>>>(lnx, x_proj_w, dbl);
  // 3) dt_proj + softplus: 512 * 16 = 8192 waves
  k_dtproj<<<MGRP * 16 / 8, 256, 0, stream>>>(dbl, dt_proj_w, dt_proj_b, delta);
  // 4) scan pass A
  k_scanA<<<BSZ * NCHUNK * DMODEL / 256, 256, 0, stream>>>(delta, lnx, dbl, A_log, cP, cH);
  // 5) scan pass B
  k_scanB<<<BSZ * DMODEL * NSTATE / 256, 256, 0, stream>>>(cP, cH, cH0);
  // 6) scan pass C: replay with carries, fused +x*D and gelu
  k_scanC<<<BSZ * NCHUNK * DMODEL / 256, 256, 0, stream>>>(delta, lnx, dbl, A_log,
                                                           D_skip, cH0, yg);
  // 7) GLU (both halves) + residual: 512 * 16 = 8192 waves
  k_glu<<<MGRP * 16 / 8, 256, 0, stream>>>(yg, glu_w, glu_b, x, out);
}
